// DiffFeatureMatcher_34540126994850
// MI455X (gfx1250) — matrix operand order repaired and checked by reading, not compiled, not run on hardware
//
#include <hip/hip_runtime.h>
#include <math.h>

typedef __attribute__((ext_vector_type(16))) _Float16 v16h;
typedef __attribute__((ext_vector_type(8)))  float    v8f;
typedef __attribute__((ext_vector_type(4)))  unsigned int v4u;
typedef __attribute__((ext_vector_type(4)))  int      v4i;
typedef __attribute__((ext_vector_type(8)))  int      v8i;

#define H_ 256
#define W_ 320
#define C_ 128
#define N_ 1024
#define HW_ (H_ * W_)             /* 81920 */
#define CHUNKS 16
#define CHUNK_COLS (HW_ / CHUNKS) /* 5120 */
#define TILE_COLS 128
#define ITERS (CHUNK_COLS / TILE_COLS) /* 40 */
#define BSTRIDE 144               /* halfs; 288B col stride (TDM pad: 8 dwords per 256B) */

// ---------------------------------------------------------------------------
// Kernel 1: convert tgt feature map to f16 *transposed* ([m][k], column
// features contiguous) and compute per-column squared norms. The 21 MB f16
// copy stays resident in the 192 MB L2 for the 64 row-tile re-reads.
// ---------------------------------------------------------------------------
__launch_bounds__(256)
__global__ void prep_tgt(const float* __restrict__ tgt,
                         _Float16* __restrict__ tgt16T,
                         float* __restrict__ tgt_sq) {
    int m = blockIdx.x * blockDim.x + threadIdx.x;
    if (m >= HW_) return;
    float sq = 0.0f;
    _Float16* row = tgt16T + (size_t)m * C_;
#pragma unroll
    for (int k8 = 0; k8 < C_; k8 += 8) {
        union { uint4 q; _Float16 h[8]; } u;
#pragma unroll
        for (int j = 0; j < 8; ++j) {
            float f = tgt[(size_t)(k8 + j) * HW_ + m]; // coalesced across threads
            u.h[j] = (_Float16)f;
            sq = fmaf(f, f, sq);
        }
        *(uint4*)(row + k8) = u.q; // 16B store, 256B/thread contiguous
    }
    tgt_sq[m] = sq;
}

// ---------------------------------------------------------------------------
// Kernel 2: keypoint indices (rintf = round-half-even like jnp.round),
// gather kp features -> f16 (row-major [n][k]), kp norms, gt indices.
// ---------------------------------------------------------------------------
__launch_bounds__(256)
__global__ void prep_kp(const float* __restrict__ src,
                        const float* __restrict__ src_kp,
                        const float* __restrict__ gt_kp,
                        _Float16* __restrict__ kp16,
                        float* __restrict__ kp_sq,
                        int* __restrict__ src_idx,
                        int* __restrict__ gt_idx) {
    int n = blockIdx.x * blockDim.x + threadIdx.x;
    if (n >= N_) return;

    float h = rintf(src_kp[n * 2 + 0]);
    h = fminf(fmaxf(h, 0.0f), (float)(H_ - 1));
    float w = rintf(src_kp[n * 2 + 1]);
    w = fminf(fmaxf(w, 0.0f), (float)(W_ - 1));
    int idx = (int)h * W_ + (int)w;
    src_idx[n] = idx;

    float gh = rintf(gt_kp[n * 2 + 0]);
    gh = fminf(fmaxf(gh, 0.0f), (float)(H_ - 1));
    float gw = rintf(gt_kp[n * 2 + 1]);
    gw = fminf(fmaxf(gw, 0.0f), (float)(W_ - 1));
    gt_idx[n] = (int)gh * W_ + (int)gw;

    float sq = 0.0f;
#pragma unroll 4
    for (int k = 0; k < C_; ++k) {
        float f = src[(size_t)k * HW_ + idx];
        kp16[(size_t)n * C_ + k] = (_Float16)f;
        sq = fmaf(f, f, sq);
    }
    kp_sq[n] = sq;
}

// ---------------------------------------------------------------------------
// TDM descriptor issue: one tensor_load_to_lds moves a full 128x128 f16 tile
// (2D tensor: data_size=2B, tile_dim0=tile_dim1=128, dim0_stride=128) from
// global to LDS, inserting 8 dwords (32B) of LDS padding every 256B row
// (pad_interval code 5, pad_amount code 7) -> 288B column stride in LDS.
// ---------------------------------------------------------------------------
__device__ __forceinline__ void tdm_load_tile(const _Float16* gsrc, unsigned ldsAddr) {
    unsigned long long ga = (unsigned long long)(uintptr_t)gsrc;
    // D# group 0: count=1 | lds_addr | global_addr[56:0] | type=2
    v4u g0 = { 1u,
               ldsAddr,
               (unsigned)(ga & 0xFFFFFFFFu),
               ((unsigned)((ga >> 32) & 0x01FFFFFFu)) | 0x80000000u };
    // D# group 1: wg_mask=0, data_size=1(2B), pad_enable=1, pad_interval=5
    // (256B), pad_amount=7 (8 dwords); tensor_dim0=128, tensor_dim1=128,
    // tile_dim0=128, tile_dim1=128, tile_dim2=0, dim0_stride=128.
    v8i g1 = { (int)0x0F510000,
               (int)(128u << 16),   /* tensor_dim0[15:0] in [31:16] */
               (int)(128u << 16),   /* tensor_dim1[15:0] in [31:16] */
               (int)(128u << 16),   /* tile_dim0 in [31:16] */
               128,                 /* tile_dim1 */
               128,                 /* tensor_dim0_stride[31:0] */
               0, 0 };
    v4i z4 = { 0, 0, 0, 0 };
#if defined(__clang_major__) && __clang_major__ >= 23
    v8i z8 = { 0, 0, 0, 0, 0, 0, 0, 0 };
    __builtin_amdgcn_tensor_load_to_lds(g0, g1, z4, z4, z8, 0);
#else
    __builtin_amdgcn_tensor_load_to_lds(g0, g1, z4, z4, 0);
#endif
}

// ---------------------------------------------------------------------------
// Kernel 3: fused GEMM (WMMA f16 -> f32 acc) + exp + per-row partial sums.
// Grid: (CHUNKS, N_/16). Block: 256 threads = 8 wave32; wave w owns columns
// [w*16, w*16+16) of each 128-column B tile. B tiles are DMA'd by the Tensor
// Data Mover into double-buffered LDS, overlapped with the WMMA chain.
// ---------------------------------------------------------------------------
__launch_bounds__(256)
__global__ void match_main(const _Float16* __restrict__ kp16,
                           const _Float16* __restrict__ tgt16T,
                           const float* __restrict__ kp_sq,
                           const float* __restrict__ tgt_sq,
                           const float* __restrict__ sigp,
                           float* __restrict__ SP) {
    __shared__ _Float16 A_s[16 * C_];                                  // 4 KB
    __shared__ __attribute__((aligned(32))) _Float16 B_d[2][TILE_COLS * BSTRIDE]; // 2x36 KB
    __shared__ float rowAccW[8][16];

    const int t    = threadIdx.x;
    const int wave = t >> 5;
    const int lane = t & 31;
    const int half = (lane >> 4) & 1;
    const int l16  = lane & 15;
    const int rowBase = blockIdx.y * 16;
    const int chunk   = blockIdx.x;
    const float sigma = sigp[0];

    // LDS byte addresses of the two B buffers (generic ptr low 32 bits = LDS
    // offset per the aperture truncation rule, ISA 10.2).
    const unsigned ldsB[2] = { (unsigned)(uintptr_t)(void*)&B_d[0][0],
                               (unsigned)(uintptr_t)(void*)&B_d[1][0] };

    // Kick off the TDM for the first tile while we stage A.
    if (wave == 0)
        tdm_load_tile(tgt16T + (size_t)(chunk * CHUNK_COLS) * C_, ldsB[0]);

    // Load the block's 16x128 A tile (4 KB): one uint4 per thread.
    {
        const uint4* s4 = (const uint4*)(kp16 + (size_t)rowBase * C_);
        ((uint4*)A_s)[t] = s4[t];
    }

    // Per-lane row constants: C/D VGPR r holds row (r + half*8).
    float ksq[8], rowAcc[8];
#pragma unroll
    for (int r = 0; r < 8; ++r) {
        ksq[r]    = kp_sq[rowBase + half * 8 + r];
        rowAcc[r] = 0.0f;
    }

    if (wave == 0) __builtin_amdgcn_s_wait_tensorcnt(0);
    __syncthreads();

    // Build the 4 A fragments once (A is loop-invariant). A layout (matches B):
    // lane half selects K run {0..15} vs {16..31}; the run is
    // K-contiguous -> two 16B LDS vector loads per fragment.
    union Frag { v16h v; uint4 q[2]; };
    Frag afr[4];
#pragma unroll
    for (int kk = 0; kk < 4; ++kk) {
        const _Float16* ap = A_s + l16 * C_ + half * 16 + kk * 32;
        afr[kk].q[0] = *(const uint4*)(ap);
        afr[kk].q[1] = *(const uint4*)(ap + 8);
    }

    const int cb = wave * 16; // this wave's local column base

    for (int it = 0; it < ITERS; ++it) {
        const int colBase = chunk * CHUNK_COLS + it * TILE_COLS;
        // DMA the next tile into the other buffer while we compute this one.
        // (Buffer (it+1)&1 was consumed at it-1; barrier below fenced it.)
        if (wave == 0 && it + 1 < ITERS)
            tdm_load_tile(tgt16T + (size_t)(colBase + TILE_COLS) * C_,
                          ldsB[(it + 1) & 1]);

        const _Float16* bcol = &B_d[it & 1][0] + (cb + l16) * BSTRIDE;

        v8f acc = {};
#pragma unroll
        for (int kk = 0; kk < 4; ++kk) {
            // B fragment: lane half selects K run {0..15} vs {16..31}; the run
            // is K-contiguous in the col-major LDS tile -> one 32B-aligned
            // v16h load (BSTRIDE*2 = 288, kbase*2 = 32*(half+2kk)).
            v16h b = *(const v16h*)(bcol + half * 16 + kk * 32);
            acc = __builtin_amdgcn_wmma_f32_16x16x32_f16(
                      false, afr[kk].v, false, b, (short)0, acc, false, false);
        }

        // Epilogue: dist -> exp; per-lane (per-column) partial sums. The
        // cross-lane reduction is hoisted out of the loop entirely.
        float tq = tgt_sq[colBase + cb + l16];
#pragma unroll
        for (int r = 0; r < 8; ++r)
            rowAcc[r] += __expf(-sigma * (ksq[r] + tq - 2.0f * acc[r]));

        if (wave == 0) __builtin_amdgcn_s_wait_tensorcnt(0);
        __syncthreads();
    }

    // One butterfly reduction across the 16 columns of each lane-half.
#pragma unroll
    for (int r = 0; r < 8; ++r) {
#pragma unroll
        for (int m = 1; m < 16; m <<= 1) rowAcc[r] += __shfl_xor(rowAcc[r], m, 32);
    }
    if (l16 == 0) {
#pragma unroll
        for (int r = 0; r < 8; ++r) rowAccW[wave][half * 8 + r] = rowAcc[r];
    }
    __syncthreads();
    if (t < 16) {
        float s = 0.0f;
#pragma unroll
        for (int w2 = 0; w2 < 8; ++w2) s += rowAccW[w2][t];
        SP[chunk * N_ + rowBase + t] = s; // deterministic partial per chunk
    }
}

// ---------------------------------------------------------------------------
// Kernel 4: reduce chunk partials, exact fp32 numerator at gt column, loss.
// ---------------------------------------------------------------------------
__launch_bounds__(256)
__global__ void finalize(const float* __restrict__ SP,
                         const float* __restrict__ src,
                         const float* __restrict__ tgt,
                         const float* __restrict__ kp_sq,
                         const float* __restrict__ tgt_sq,
                         const int* __restrict__ src_idx,
                         const int* __restrict__ gt_idx,
                         const float* __restrict__ sigp,
                         float* __restrict__ out) {
    __shared__ float red[256];
    const int t = threadIdx.x;
    const float sigma = sigp[0];
    float acc = 0.0f;
    for (int n = t; n < N_; n += 256) {
        float S = 0.0f;
#pragma unroll
        for (int cch = 0; cch < CHUNKS; ++cch) S += SP[cch * N_ + n];
        int g  = gt_idx[n];
        int si = src_idx[n];
        float dot = 0.0f;
#pragma unroll 4
        for (int k = 0; k < C_; ++k)
            dot = fmaf(src[(size_t)k * HW_ + si], tgt[(size_t)k * HW_ + g], dot);
        float E = __expf(-sigma * (kp_sq[n] + tgt_sq[g] - 2.0f * dot));
        acc += -logf(1e-10f + E / S);
    }
    red[t] = acc;
    __syncthreads();
    for (int s = 128; s > 0; s >>= 1) {
        if (t < s) red[t] += red[t + s];
        __syncthreads();
    }
    if (t == 0) out[0] = red[0] / (float)N_;
}

// ---------------------------------------------------------------------------
extern "C" void kernel_launch(void* const* d_in, const int* in_sizes, int n_in,
                              void* d_out, int out_size, void* d_ws, size_t ws_size,
                              hipStream_t stream) {
    const float* src    = (const float*)d_in[0]; // (1,128,256,320)
    const float* tgt    = (const float*)d_in[1]; // (1,128,256,320)
    const float* src_kp = (const float*)d_in[2]; // (1,1024,2)
    const float* gt_kp  = (const float*)d_in[3]; // (1,1024,2)
    const float* sig    = (const float*)d_in[4]; // (1,)

    char* ws = (char*)d_ws;
    size_t off = 0;
    auto alloc = [&](size_t bytes) -> char* {
        char* p = ws + off;
        off = (off + bytes + 255) & ~(size_t)255;
        return p;
    };
    _Float16* tgt16T = (_Float16*)alloc((size_t)HW_ * C_ * sizeof(_Float16)); // 21 MB, [m][k]
    _Float16* kp16   = (_Float16*)alloc((size_t)N_ * C_ * sizeof(_Float16));  // 256 KB, [n][k]
    float* tgt_sq    = (float*)alloc((size_t)HW_ * sizeof(float));
    float* kp_sq     = (float*)alloc((size_t)N_ * sizeof(float));
    int*   sidx      = (int*)alloc((size_t)N_ * sizeof(int));
    int*   gidx      = (int*)alloc((size_t)N_ * sizeof(int));
    float* SP        = (float*)alloc((size_t)CHUNKS * N_ * sizeof(float));

    prep_tgt<<<HW_ / 256, 256, 0, stream>>>(tgt, tgt16T, tgt_sq);
    prep_kp<<<N_ / 256, 256, 0, stream>>>(src, src_kp, gt_kp, kp16, kp_sq, sidx, gidx);
    match_main<<<dim3(CHUNKS, N_ / 16), 256, 0, stream>>>(kp16, tgt16T, kp_sq, tgt_sq, sig, SP);
    finalize<<<1, 256, 0, stream>>>(SP, src, tgt, kp_sq, tgt_sq, sidx, gidx, sig, (float*)d_out);
}
